// EncoderTransformer_61194694033513
// MI455X (gfx1250) — matrix operand order repaired and checked by reading, not compiled, not run on hardware
//
#include <hip/hip_runtime.h>

// Problem constants (match reference)
#define B_  16
#define N_  2048
#define D_  128
#define E_  256
#define L_  3

typedef unsigned short u16;
typedef unsigned int   u32;

typedef __bf16 v16bf __attribute__((ext_vector_type(16)));
typedef float  v8f   __attribute__((ext_vector_type(8)));
typedef int    v4i   __attribute__((ext_vector_type(4)));

struct __attribute__((aligned(16))) U128 { u32 x, y, z, w; };
union Frag16 { v16bf v; U128 q[2]; };

__device__ __forceinline__ u16 f2bf(float f) {
  u32 u = __float_as_uint(f);
  u32 r = u + 0x7fffu + ((u >> 16) & 1u);   // round-to-nearest-even
  return (u16)(r >> 16);
}

// ---------------------------------------------------------------------------
// Async global->LDS staging (CDNA5 GLOBAL_LOAD_ASYNC_TO_LDS_B128, ASYNCcnt).
// Builtin signature (from hipcc diagnostic): (AS1 v4i* src, AS3 v4i* dst,
// imm offset, imm cpol). Falls back to a plain 16B copy if not declared.
// ---------------------------------------------------------------------------
#if defined(__has_builtin)
#if __has_builtin(__builtin_amdgcn_global_load_async_to_lds_b128)
#define HAVE_ASYNC 1
#endif
#endif
#ifndef HAVE_ASYNC
#define HAVE_ASYNC 0
#endif

__device__ __forceinline__ void cp16(u16* l, const u16* g) {
#if HAVE_ASYNC
  __builtin_amdgcn_global_load_async_to_lds_b128(
      (__attribute__((address_space(1))) v4i*)(v4i*)const_cast<u16*>(g),
      (__attribute__((address_space(3))) v4i*)(v4i*)l, 0, 0);
#else
  *(U128*)l = *(const U128*)g;
#endif
}

__device__ __forceinline__ void async_wait() {
#if HAVE_ASYNC
#if __has_builtin(__builtin_amdgcn_s_wait_asynccnt)
  __builtin_amdgcn_s_wait_asynccnt(0);
#else
  asm volatile("s_wait_asynccnt 0x0" ::: "memory");
#endif
#endif
}

// ---------------------------------------------------------------------------
// Weight prep: W f32 [K][256] row-major  ->  Wt bf16 [256][K] (transposed),
// done once per launch so the GEMM hot loop sees pure bf16 b128 copies.
// ---------------------------------------------------------------------------
__global__ void prep_w(const float* __restrict__ W, u16* __restrict__ Wt, int K) {
  int i = blockIdx.x * blockDim.x + threadIdx.x;
  int n = K * E_;
  int stride = gridDim.x * blockDim.x;
  for (; i < n; i += stride) {
    int nCol = i / K, k = i - nCol * K;
    Wt[i] = f2bf(W[(size_t)k * E_ + nCol]);
  }
}

// ---------------------------------------------------------------------------
// GEMM: C[M,256] = act( A_bf16[M,K] @ W[K,256] + bias ), Wt pre-transposed
// bf16 [256][K]. Block tile 128x128, 8 waves of 64x32, K-step 32,
// double-buffered LDS staged with async b128 loads, v_wmma_f32_16x16x32_bf16.
// ---------------------------------------------------------------------------
__global__ __launch_bounds__(256) void gemm_bf16(
    const u16*  __restrict__ A, const u16* __restrict__ Wt,
    const float* __restrict__ bias, float* __restrict__ outF,
    u16* __restrict__ outB, int K, int doRelu)
{
  __shared__ __attribute__((aligned(16))) u16 sA[2][128 * 32];  // A tile [m][k]
  __shared__ __attribute__((aligned(16))) u16 sB[2][128 * 32];  // W^T tile [n][k]

  const int t      = threadIdx.x;
  const int lane   = t & 31;
  const int wave   = t >> 5;
  const int laneR  = lane & 15;
  const int laneHi = lane >> 4;
  const int wm     = wave & 1;     // 2 waves along M (64 rows each)
  const int wn     = wave >> 1;    // 4 waves along N (32 cols each)
  const int m0     = blockIdx.x * 128;
  const int n0     = blockIdx.y * 128;

  v8f acc[4][2];
  for (int i = 0; i < 4; i++)
    for (int j = 0; j < 2; j++)
      for (int r = 0; r < 8; r++) acc[i][j][r] = 0.f;

  const int rowS  = t >> 1;        // 128 rows, 2 threads/row
  const int halfS = t & 1;         // 16-u16 half of a 32-u16 row

  auto stage = [&](int p, int kb) {
    const u16* ga = A  + (size_t)(m0 + rowS) * K + kb + halfS * 16;
    const u16* gw = Wt + (size_t)(n0 + rowS) * K + kb + halfS * 16;
    u16* la = &sA[p][rowS * 32 + halfS * 16];
    u16* lw = &sB[p][rowS * 32 + halfS * 16];
    cp16(la, ga); cp16(la + 8, ga + 8);
    cp16(lw, gw); cp16(lw + 8, gw + 8);
  };

  stage(0, 0);
  async_wait();
  __syncthreads();

  const int nk = K >> 5;
  for (int i = 0; i < nk; i++) {
    const int p = i & 1;
    if (i + 1 < nk) stage(p ^ 1, (i + 1) * 32);   // async prefetch next k-block

    Frag16 a[4], b[2];
    for (int mt = 0; mt < 4; mt++) {
      int r = wm * 64 + mt * 16 + laneR;
      a[mt].q[0] = *(const U128*)&sA[p][r * 32 + laneHi * 8];       // K 0..7 / 8..15
      a[mt].q[1] = *(const U128*)&sA[p][r * 32 + 16 + laneHi * 8];  // K 16..23 / 24..31
    }
    for (int nt = 0; nt < 2; nt++) {
      int c = wn * 32 + nt * 16 + laneR;
      const U128* pb = (const U128*)&sB[p][c * 32 + laneHi * 8];    // K laneHi*8..+7 / 16+laneHi*8..+7
      b[nt].q[0] = pb[0]; b[nt].q[1] = pb[2];
    }
    for (int mt = 0; mt < 4; mt++)
      for (int nt = 0; nt < 2; nt++)
        acc[mt][nt] = __builtin_amdgcn_wmma_f32_16x16x32_bf16(
            false, a[mt].v, false, b[nt].v, (short)0, acc[mt][nt], false, false);

    if (i + 1 < nk) {
      async_wait();
      __syncthreads();
    }
  }

  for (int mt = 0; mt < 4; mt++)
    for (int nt = 0; nt < 2; nt++) {
      int col = n0 + wn * 32 + nt * 16 + laneR;
      float bv = bias ? bias[col] : 0.f;
      for (int r = 0; r < 8; r++) {
        int row = m0 + wm * 64 + mt * 16 + laneHi * 8 + r;
        float v = acc[mt][nt][r] + bv;
        if (doRelu) v = v > 0.f ? v : 0.f;
        size_t idx = (size_t)row * E_ + col;
        if (outF) outF[idx] = v;
        if (outB) outB[idx] = f2bf(v);
      }
    }
}

// ---------------------------------------------------------------------------
// Flash-style attention (relu scores, no softmax):
//   O[q0..q0+63, :] = sum_over_64-key-blocks relu(Q Kblk^T) Vblk
// Q/K staged with async b128 loads; V staged transposed via register
// pair-packing (b128 global reads -> b32 LDS writes). S round-trips through
// LDS in bf16 to re-layout C->A fragments. Never materializes [N,N].
// ---------------------------------------------------------------------------
__global__ __launch_bounds__(256) void attn_relu(
    const u16* __restrict__ Q, const u16* __restrict__ Km,
    const u16* __restrict__ V, float* __restrict__ O)
{
  extern __shared__ u16 smem[];
  u16* sQ  = smem;                 // [64][256]
  u16* sK  = sQ  + 64 * 256;       // [64][256]
  u16* sVt = sK  + 64 * 256;       // [256][64]  (V^T)
  u16* sS  = sVt + 256 * 64;       // [64][64]

  const int t      = threadIdx.x;
  const int lane   = t & 31;
  const int wave   = t >> 5;
  const int laneR  = lane & 15;
  const int laneHi = lane >> 4;
  const int wm     = wave & 1;     // 2 waves along query rows (32 each)
  const int wn     = wave >> 1;    // 4 waves: 16 key-cols (S) / 64 E-cols (O)

  const int b  = blockIdx.y;
  const int q0 = blockIdx.x * 64;
  const u16* Qb = Q  + ((size_t)b * N_ + q0) * E_;
  const u16* Kb = Km + (size_t)b * N_ * E_;
  const u16* Vb = V  + (size_t)b * N_ * E_;
  float*     Ob = O  + ((size_t)b * N_ + q0) * E_;

  const int rowC = t >> 2, segC = t & 3;   // copy mapping: 4 threads/row

  { // stage Q once (32 KB, async)
    const u16* src = Qb + (size_t)rowC * E_ + segC * 64;
    u16* dst = &sQ[rowC * 256 + segC * 64];
    for (int i = 0; i < 8; i++) cp16(dst + i * 8, src + i * 8);
  }

  v8f o[2][4];
  for (int i = 0; i < 2; i++)
    for (int j = 0; j < 4; j++)
      for (int r = 0; r < 8; r++) o[i][j][r] = 0.f;

  const int jl = (t & 31) * 2;         // V transpose mapping: row pair
  const int es = (t >> 5) * 32;        // 32-wide E segment

  for (int j0 = 0; j0 < N_; j0 += 64) {
    __syncthreads();
    { // stage K block (async)
      const u16* src = Kb + (size_t)(j0 + rowC) * E_ + segC * 64;
      u16* dst = &sK[rowC * 256 + segC * 64];
      for (int i = 0; i < 8; i++) cp16(dst + i * 8, src + i * 8);
    }
    { // stage V block transposed: b128 reads, packed b32 LDS writes
      U128 r0[4], r1[4];
      const U128* s0 = (const U128*)(Vb + (size_t)(j0 + jl) * E_ + es);
      const U128* s1 = (const U128*)(Vb + (size_t)(j0 + jl + 1) * E_ + es);
      for (int i = 0; i < 4; i++) { r0[i] = s0[i]; r1[i] = s1[i]; }
      const u16* a0 = (const u16*)r0;
      const u16* a1 = (const u16*)r1;
      for (int e = 0; e < 32; e++) {
        u32 pack = (u32)a0[e] | ((u32)a1[e] << 16);
        *(u32*)&sVt[(es + e) * 64 + jl] = pack;
      }
    }
    async_wait();
    __syncthreads();

    // S = Q * Kblk^T  (wave tile: 32 rows x 16 keys)
    v8f s[2];
    for (int mt = 0; mt < 2; mt++)
      for (int r = 0; r < 8; r++) s[mt][r] = 0.f;
    for (int e = 0; e < E_; e += 32) {
      Frag16 a[2], bk;
      for (int mt = 0; mt < 2; mt++) {
        int r = wm * 32 + mt * 16 + laneR;
        a[mt].q[0] = *(const U128*)&sQ[r * 256 + e + laneHi * 8];
        a[mt].q[1] = *(const U128*)&sQ[r * 256 + e + 16 + laneHi * 8];
      }
      {
        int c = wn * 16 + laneR;                 // local key index
        const U128* p = (const U128*)&sK[c * 256 + e + laneHi * 8];
        bk.q[0] = p[0]; bk.q[1] = p[2];
      }
      for (int mt = 0; mt < 2; mt++)
        s[mt] = __builtin_amdgcn_wmma_f32_16x16x32_bf16(
            false, a[mt].v, false, bk.v, (short)0, s[mt], false, false);
    }

    // relu + bf16 -> LDS (re-layout C -> A fragments)
    for (int mt = 0; mt < 2; mt++)
      for (int r = 0; r < 8; r++) {
        int row = wm * 32 + mt * 16 + laneHi * 8 + r;
        int col = wn * 16 + laneR;
        float v = s[mt][r];
        sS[row * 64 + col] = f2bf(v > 0.f ? v : 0.f);
      }
    __syncthreads();

    // O += S * Vblk  (wave tile: 32 rows x 64 E-cols)
    for (int kk = 0; kk < 64; kk += 32) {
      Frag16 a[2], bv[4];
      for (int mt = 0; mt < 2; mt++) {
        int r = wm * 32 + mt * 16 + laneR;
        a[mt].q[0] = *(const U128*)&sS[r * 64 + kk + laneHi * 8];
        a[mt].q[1] = *(const U128*)&sS[r * 64 + kk + 16 + laneHi * 8];
      }
      for (int nt = 0; nt < 4; nt++) {
        int e = wn * 64 + nt * 16 + laneR;
        const U128* p = (const U128*)&sVt[e * 64 + kk + laneHi * 8];
        bv[nt].q[0] = p[0]; bv[nt].q[1] = p[2];
      }
      for (int mt = 0; mt < 2; mt++)
        for (int nt = 0; nt < 4; nt++)
          o[mt][nt] = __builtin_amdgcn_wmma_f32_16x16x32_bf16(
              false, a[mt].v, false, bv[nt].v, (short)0, o[mt][nt], false, false);
    }
  }

  for (int mt = 0; mt < 2; mt++)
    for (int nt = 0; nt < 4; nt++)
      for (int r = 0; r < 8; r++) {
        int row = wm * 32 + mt * 16 + laneHi * 8 + r;
        int col = wn * 64 + nt * 16 + laneR;
        Ob[(size_t)row * E_ + col] = o[mt][nt][r];
      }
}

// ---------------------------------------------------------------------------
// Y = LayerNorm(X + R) * g + beta ; writes f32 and bf16 copies. One row/block.
// ---------------------------------------------------------------------------
__global__ __launch_bounds__(256) void ln_res(
    const float* __restrict__ X, const float* __restrict__ R,
    const float* __restrict__ g, const float* __restrict__ beta,
    float* __restrict__ Yf, u16* __restrict__ Yb)
{
  __shared__ float red[256];
  const int t = threadIdx.x;
  const size_t row = blockIdx.x;
  float x = X[row * E_ + t] + R[row * E_ + t];
  red[t] = x; __syncthreads();
  for (int s2 = 128; s2 > 0; s2 >>= 1) { if (t < s2) red[t] += red[t + s2]; __syncthreads(); }
  float mu = red[0] * (1.f / E_);
  __syncthreads();
  float d = x - mu;
  red[t] = d * d; __syncthreads();
  for (int s2 = 128; s2 > 0; s2 >>= 1) { if (t < s2) red[t] += red[t + s2]; __syncthreads(); }
  float var = red[0] * (1.f / E_);
  float y = d * rsqrtf(var + 1e-5f) * g[t] + beta[t];
  Yf[row * E_ + t] = y;
  Yb[row * E_ + t] = f2bf(y);
}

__global__ void cvt_bf16(const float* __restrict__ x, u16* __restrict__ y, int n) {
  int i = blockIdx.x * blockDim.x + threadIdx.x;
  int stride = gridDim.x * blockDim.x;
  for (; i < n; i += stride) y[i] = f2bf(x[i]);
}

// Deterministic two-stage final reduction: out[b] = H[b].flat dot Wout + b_out
__global__ __launch_bounds__(256) void red1(const float* __restrict__ H,
    const float* __restrict__ Wout, float* __restrict__ part)
{
  __shared__ float red[256];
  int b = blockIdx.y, c = blockIdx.x, t = threadIdx.x;
  size_t base = (size_t)b * N_ * E_ + (size_t)c * 16384;
  size_t wb = (size_t)c * 16384;
  float s = 0.f;
  for (int i = t; i < 16384; i += 256) s += H[base + i] * Wout[wb + i];
  red[t] = s; __syncthreads();
  for (int s2 = 128; s2 > 0; s2 >>= 1) { if (t < s2) red[t] += red[t + s2]; __syncthreads(); }
  if (t == 0) part[b * 32 + c] = red[0];
}

__global__ void red2(const float* __restrict__ part, const float* __restrict__ b_out,
                     float* __restrict__ out)
{
  int t = threadIdx.x;
  if (t < B_) {
    float s = b_out[0];
    for (int i = 0; i < 32; i++) s += part[t * 32 + i];
    out[t] = s;
  }
}

extern "C" void kernel_launch(void* const* d_in, const int* in_sizes, int n_in,
                              void* d_out, int out_size, void* d_ws, size_t ws_size,
                              hipStream_t stream) {
  const float* xs    = (const float*)d_in[0];
  const float* Win   = (const float*)d_in[1];
  const float* b_in  = (const float*)d_in[2];
  const float* Wq    = (const float*)d_in[3];
  const float* Wk    = (const float*)d_in[4];
  const float* Wv    = (const float*)d_in[5];
  const float* g1    = (const float*)d_in[6];
  const float* be1   = (const float*)d_in[7];
  const float* W1    = (const float*)d_in[8];
  const float* bm1   = (const float*)d_in[9];
  const float* W2    = (const float*)d_in[10];
  const float* bm2   = (const float*)d_in[11];
  const float* g2    = (const float*)d_in[12];
  const float* be2   = (const float*)d_in[13];
  const float* Wout  = (const float*)d_in[14];
  const float* b_out = (const float*)d_in[15];
  float* out = (float*)d_out;

  const size_t M  = (size_t)B_ * N_;      // 32768 rows
  const size_t EE = (size_t)E_ * E_;
  char* w = (char*)d_ws;
  float* Hf   = (float*)w; w += M * E_ * 4;     // f32 activations
  u16*   Hb   = (u16*)w;   w += M * E_ * 2;     // bf16 activations
  u16*   Qb   = (u16*)w;   w += M * E_ * 2;
  u16*   Kb   = (u16*)w;   w += M * E_ * 2;
  u16*   Vb   = (u16*)w;   w += M * E_ * 2;
  u16*   Xb   = (u16*)w;   w += M * D_ * 2;     // xs in bf16
  float* T1   = (float*)w; w += M * E_ * 4;     // attention / mlp2 output
  float* part = (float*)w; w += B_ * 32 * 4;
  u16*   WinT = (u16*)w;   w += (size_t)D_ * E_ * 2;   // transposed bf16 weights
  u16*   WT   = (u16*)w;   w += (size_t)15 * EE * 2;   // [l*5 + {q,k,v,m1,m2}]

  // One-time weight transpose+convert and input conversion
  cvt_bf16<<<2048, 256, 0, stream>>>(xs, Xb, (int)(M * D_));
  prep_w<<<32, 256, 0, stream>>>(Win, WinT, D_);
  const float* wsrc[5] = { Wq, Wk, Wv, W1, W2 };
  for (int l = 0; l < L_; l++)
    for (int j = 0; j < 5; j++)
      prep_w<<<64, 256, 0, stream>>>(wsrc[j] + l * EE, WT + ((size_t)l * 5 + j) * EE, E_);

  dim3 gg((unsigned)(M / 128), E_ / 128);

  // H = xs @ Win + b_in
  gemm_bf16<<<gg, 256, 0, stream>>>(Xb, WinT, b_in, Hf, Hb, D_, 0);

  const size_t shm = (size_t)(64 * 256 * 2 + 256 * 64 + 64 * 64) * sizeof(u16); // 104 KB

  for (int l = 0; l < L_; l++) {
    const u16* WqT = WT + ((size_t)l * 5 + 0) * EE;
    const u16* WkT = WT + ((size_t)l * 5 + 1) * EE;
    const u16* WvT = WT + ((size_t)l * 5 + 2) * EE;
    const u16* W1T = WT + ((size_t)l * 5 + 3) * EE;
    const u16* W2T = WT + ((size_t)l * 5 + 4) * EE;

    gemm_bf16<<<gg, 256, 0, stream>>>(Hb, WqT, nullptr, nullptr, Qb, E_, 0);
    gemm_bf16<<<gg, 256, 0, stream>>>(Hb, WkT, nullptr, nullptr, Kb, E_, 0);
    gemm_bf16<<<gg, 256, 0, stream>>>(Hb, WvT, nullptr, nullptr, Vb, E_, 0);
    attn_relu<<<dim3(N_ / 64, B_), 256, shm, stream>>>(Qb, Kb, Vb, T1);
    ln_res<<<(unsigned)M, 256, 0, stream>>>(Hf, T1, g1 + l * E_, be1 + l * E_, Hf, Hb);
    gemm_bf16<<<gg, 256, 0, stream>>>(Hb, W1T, bm1 + l * E_, nullptr, Qb, E_, 1); // reuse Qb
    gemm_bf16<<<gg, 256, 0, stream>>>(Qb, W2T, bm2 + l * E_, T1, nullptr, E_, 0);
    ln_res<<<(unsigned)M, 256, 0, stream>>>(Hf, T1, g2 + l * E_, be2 + l * E_, Hf, Hb);
  }

  red1<<<dim3(32, B_), 256, 0, stream>>>(Hf, Wout, part);
  red2<<<1, 64, 0, stream>>>(part, b_out, out);
}
